// TDrumorGAT_34574486733591
// MI455X (gfx1250) — matrix operand order repaired and checked by reading, not compiled, not run on hardware
//
#include <hip/hip_runtime.h>

// ---------------------------------------------------------------------------
// TDrumorGAT on gfx1250: bf16 WMMA GEMMs (4-way N-blocked per wave) +
// memory-bound edge softmax/scatter.
// ---------------------------------------------------------------------------

#define NHEAD 4
#define CH 64         // channels per head (256/4)
#define FW 256        // feature width of both layers' outputs

typedef __bf16 bf16_t;
typedef __attribute__((ext_vector_type(16))) __bf16 v16bf;
typedef __attribute__((ext_vector_type(8)))  __bf16 v8bf;
typedef __attribute__((ext_vector_type(8)))  float  v8f;

__device__ __forceinline__ unsigned short f32_to_bf16_rte(float f) {
  unsigned u = __float_as_uint(f);
  unsigned r = (u + 0x7FFFu + ((u >> 16) & 1u)) >> 16;
  return (unsigned short)r;
}

__device__ __forceinline__ void atomicMaxF(float* addr, float v) {
  // Ordered-int trick: works with -inf init, mixed signs.
  if (v >= 0.0f) atomicMax((int*)addr, __float_as_int(v));
  else           atomicMin((unsigned int*)addr, __float_as_uint(v));
}

// --------------------------- utility kernels -------------------------------

__global__ void fill_f32_kernel(float* p, float v, long count) {
  long t = (long)blockIdx.x * blockDim.x + threadIdx.x;
  if (t < count) p[t] = v;
}

__global__ void conv_bf16_kernel(const float* __restrict__ src,
                                 unsigned short* __restrict__ dst, long count) {
  long t = (long)blockIdx.x * blockDim.x + threadIdx.x;
  if (t < count) dst[t] = f32_to_bf16_rte(src[t]);
}

// W: rows x cols (row-major) -> WT: cols x rows (row-major over K=rows), bf16
__global__ void transpose_bf16_kernel(const float* __restrict__ W,
                                      unsigned short* __restrict__ WT,
                                      int rows, int cols) {
  long t = (long)blockIdx.x * blockDim.x + threadIdx.x;
  if (t >= (long)rows * cols) return;
  int r = (int)(t / cols), c = (int)(t % cols);
  WT[(long)c * rows + r] = f32_to_bf16_rte(W[t]);
}

// --------------------------- WMMA GEMM -------------------------------------
// C[M x Nout] = A[M x K] (bf16, row-major) @ BT[Nout x K]^T (bf16, row-major)
// One wave32 computes a 16x64 output strip (4 N-tiles): the A fragment is
// loaded once per K-step and reused by 4 v_wmma_f32_16x16x32_bf16, cutting
// load bytes per WMMA from 64B to 40B per lane. M mult of 16, Nout mult of
// 64, K mult of 32.
__global__ void gemm_bf16_wmma_kernel(const bf16_t* __restrict__ A,
                                      const bf16_t* __restrict__ BT,
                                      float* __restrict__ C,
                                      int M, int K, int Nout) {
  const int lane = threadIdx.x & 31;
  const int wave = threadIdx.x >> 5;
  const int ngrp = Nout >> 6;                       // groups of 4 N-tiles
  const int tile = blockIdx.x * (blockDim.x >> 5) + wave;
  if (tile >= (M >> 4) * ngrp) return;              // wave-uniform guard
  const int mt = tile / ngrp, ng = tile % ngrp;
  const int r = lane & 15;
  const int half = lane >> 4;
  // A fragment: lane holds row (mt*16+r); 16 contiguous K @ 16*half
  const bf16_t* ap = A  + (long)(mt * 16 + r) * K + half * 16;
  // B fragments: lane holds col (ng*64 + 16*j + r); 16 contiguous K @ 16*half
  const bf16_t* bp = BT + (long)(ng * 64 + r) * K + half * 16;
  const long bstride = (long)16 * K;
  v8f acc0 = {}, acc1 = {}, acc2 = {}, acc3 = {};
  for (int k0 = 0; k0 < K; k0 += 32) {
    v8bf alo = *(const v8bf*)(ap + k0);
    v8bf ahi = *(const v8bf*)(ap + k0 + 8);
    v16bf av = __builtin_shufflevector(alo, ahi,
        0,1,2,3,4,5,6,7,8,9,10,11,12,13,14,15);
    v16bf b0 = *(const v16bf*)(bp + k0);
    v16bf b1 = *(const v16bf*)(bp + bstride + k0);
    v16bf b2 = *(const v16bf*)(bp + 2 * bstride + k0);
    v16bf b3 = *(const v16bf*)(bp + 3 * bstride + k0);
    acc0 = __builtin_amdgcn_wmma_f32_16x16x32_bf16(false, av, false, b0, (short)0, acc0, false, false);
    acc1 = __builtin_amdgcn_wmma_f32_16x16x32_bf16(false, av, false, b1, (short)0, acc1, false, false);
    acc2 = __builtin_amdgcn_wmma_f32_16x16x32_bf16(false, av, false, b2, (short)0, acc2, false, false);
    acc3 = __builtin_amdgcn_wmma_f32_16x16x32_bf16(false, av, false, b3, (short)0, acc3, false, false);
  }
  const int mbase = mt * 16 + half * 8;
  const int nc = ng * 64 + r;
#pragma unroll
  for (int rr = 0; rr < 8; ++rr) {
    float* crow = C + (long)(mbase + rr) * Nout + nc;
    crow[0]  = acc0[rr];
    crow[16] = acc1[rr];
    crow[32] = acc2[rr];
    crow[48] = acc3[rr];
  }
}

// ------------------------ attention logits ---------------------------------
// als[n,h] = sum_c h[n, h*64+c]*a_s[h,c] ; ald likewise.
__global__ void logits_kernel(const float* __restrict__ h,
                              const float* __restrict__ a_s,
                              const float* __restrict__ a_d,
                              float* __restrict__ als, float* __restrict__ ald,
                              int n_nodes) {
  long t = (long)blockIdx.x * blockDim.x + threadIdx.x;
  if (t >= (long)n_nodes * NHEAD) return;
  int n = (int)(t >> 2), hd = (int)(t & 3);
  const float* hp = h + (long)n * FW + hd * CH;
  const float* sp = a_s + hd * CH;
  const float* dp = a_d + hd * CH;
  float ss = 0.f, sd = 0.f;
#pragma unroll 8
  for (int c = 0; c < CH; ++c) { float hv = hp[c]; ss += hv * sp[c]; sd += hv * dp[c]; }
  als[t] = ss; ald[t] = sd;
}

// --------------------- segment softmax (3 passes) --------------------------
// edges idx in [0, E): (src,dst) from edge_index; idx in [E, E+N): self loop.

__global__ void edge_logit_max_kernel(const int* __restrict__ ei,
                                      const float* __restrict__ als,
                                      const float* __restrict__ ald,
                                      float* __restrict__ ebuf,
                                      float* __restrict__ emax,
                                      int E_, int n_nodes) {
  long t = (long)blockIdx.x * blockDim.x + threadIdx.x;
  long total = (long)(E_ + n_nodes) * NHEAD;
  if (t >= total) return;
  long idx = t >> 2; int hd = (int)(t & 3);
  int src, dst;
  if (idx < E_) { src = ei[idx]; dst = ei[(long)E_ + idx]; }
  else          { src = dst = (int)(idx - E_); }
  float e = als[(long)src * NHEAD + hd] + ald[(long)dst * NHEAD + hd];
  e = (e > 0.f) ? e : 0.2f * e;                 // leaky_relu(0.2)
  ebuf[t] = e;
  atomicMaxF(&emax[(long)dst * NHEAD + hd], e);
}

__global__ void edge_exp_sum_kernel(const int* __restrict__ ei,
                                    float* __restrict__ ebuf,
                                    const float* __restrict__ emax,
                                    float* __restrict__ denom,
                                    int E_, int n_nodes) {
  long t = (long)blockIdx.x * blockDim.x + threadIdx.x;
  long total = (long)(E_ + n_nodes) * NHEAD;
  if (t >= total) return;
  long idx = t >> 2; int hd = (int)(t & 3);
  int dst = (idx < E_) ? ei[(long)E_ + idx] : (int)(idx - E_);
  float p = __expf(ebuf[t] - emax[(long)dst * NHEAD + hd]);
  ebuf[t] = p;
  atomicAdd(&denom[(long)dst * NHEAD + hd], p);
}

// wave32 per edge: lane handles 8 contiguous features (float4 x2 gather,
// 8 global f32 atomics scatter).
__global__ void edge_aggregate_kernel(const int* __restrict__ ei,
                                      const float* __restrict__ ebuf,
                                      const float* __restrict__ denom,
                                      const float* __restrict__ h,
                                      float* __restrict__ agg,
                                      int E_, int n_nodes) {
  int lane = threadIdx.x & 31;
  long warp = (long)blockIdx.x * (blockDim.x >> 5) + (threadIdx.x >> 5);
  long total = (long)E_ + n_nodes;
  if (warp >= total) return;
  long idx = warp;
  int src, dst;
  if (idx < E_) { src = ei[idx]; dst = ei[(long)E_ + idx]; }
  else          { src = dst = (int)(idx - E_); }
  int f0 = lane * 8;
  int hd = lane >> 3;
  float alpha = ebuf[idx * NHEAD + hd] /
                (denom[(long)dst * NHEAD + hd] + 1e-16f);
  const float4* hp = (const float4*)(h + (long)src * FW + f0);
  float4 v0 = hp[0], v1 = hp[1];
  float* ap = agg + (long)dst * FW + f0;
  atomicAdd(ap + 0, alpha * v0.x); atomicAdd(ap + 1, alpha * v0.y);
  atomicAdd(ap + 2, alpha * v0.z); atomicAdd(ap + 3, alpha * v0.w);
  atomicAdd(ap + 4, alpha * v1.x); atomicAdd(ap + 5, alpha * v1.y);
  atomicAdd(ap + 6, alpha * v1.z); atomicAdd(ap + 7, alpha * v1.w);
}

// ---------------- bias + BN(eval) + LayerNorm + ReLU per node --------------
__global__ void post_kernel(const float* __restrict__ agg,
                            const float* __restrict__ bias,
                            const float* __restrict__ bng, const float* __restrict__ bnb,
                            const float* __restrict__ bnm, const float* __restrict__ bnv,
                            const float* __restrict__ lng, const float* __restrict__ lnb,
                            float* __restrict__ outp) {
  int n = blockIdx.x;
  int f = threadIdx.x;  // 256 threads
  __shared__ float red[FW];
  float t = agg[(long)n * FW + f] + bias[f];
  float z = (t - bnm[f]) * (bng[f] * rsqrtf(bnv[f] + 1e-5f)) + bnb[f];
  red[f] = z; __syncthreads();
  for (int s = FW / 2; s > 0; s >>= 1) { if (f < s) red[f] += red[f + s]; __syncthreads(); }
  float mu = red[0] * (1.0f / FW); __syncthreads();
  float d = z - mu;
  red[f] = d * d; __syncthreads();
  for (int s = FW / 2; s > 0; s >>= 1) { if (f < s) red[f] += red[f + s]; __syncthreads(); }
  float var = red[0] * (1.0f / FW);
  float y = d * rsqrtf(var + 1e-5f) * lng[f] + lnb[f];
  outp[(long)n * FW + f] = fmaxf(y, 0.f);
}

// --------------------------- root extend -----------------------------------
__global__ void root_detect_kernel(const int* __restrict__ batch,
                                   int* __restrict__ root_idx, int n_nodes) {
  int n = blockIdx.x * blockDim.x + threadIdx.x;
  if (n >= n_nodes) return;
  if (n == 0 || batch[n] != batch[n - 1]) root_idx[batch[n]] = n;
}

__global__ void root_gather_kernel(const int* __restrict__ batch,
                                   const int* __restrict__ root_idx,
                                   int* __restrict__ ron, int n_nodes) {
  int n = blockIdx.x * blockDim.x + threadIdx.x;
  if (n < n_nodes) ron[n] = root_idx[batch[n]];
}

// h2in[n, 0:256] = bf16(post1[n]); h2in[n, 256:512] = xbf[root(n)]
__global__ void build_h2in_kernel(const float* __restrict__ post1,
                                  const unsigned short* __restrict__ xbf,
                                  const int* __restrict__ ron,
                                  unsigned short* __restrict__ h2in, int n_nodes) {
  long t = (long)blockIdx.x * blockDim.x + threadIdx.x;
  if (t >= (long)n_nodes * 512) return;
  int n = (int)(t >> 9), f = (int)(t & 511);
  unsigned short v;
  if (f < FW) v = f32_to_bf16_rte(post1[(long)n * FW + f]);
  else        v = xbf[(long)ron[n] * FW + (f - FW)];
  h2in[t] = v;
}

// ----------------------- final segment mean --------------------------------
__global__ void final_accum_kernel(const float* __restrict__ post2,
                                   const float* __restrict__ post1,
                                   const int* __restrict__ ron,
                                   const int* __restrict__ batch,
                                   float* __restrict__ sums,
                                   float* __restrict__ counts, int n_nodes) {
  long t = (long)blockIdx.x * blockDim.x + threadIdx.x;
  if (t >= (long)n_nodes * 512) return;
  int n = (int)(t >> 9), f = (int)(t & 511);
  float v = (f < FW) ? post2[(long)n * FW + f]
                     : post1[(long)ron[n] * FW + (f - FW)];
  atomicAdd(&sums[(long)batch[n] * 512 + f], v);
  if (f == 0) atomicAdd(&counts[batch[n]], 1.0f);
}

__global__ void final_div_kernel(float* __restrict__ sums,
                                 const float* __restrict__ counts, int Bg) {
  long t = (long)blockIdx.x * blockDim.x + threadIdx.x;
  if (t >= (long)Bg * 512) return;
  sums[t] /= fmaxf(counts[t >> 9], 1.0f);
}

// ---------------------------------------------------------------------------

extern "C" void kernel_launch(void* const* d_in, const int* in_sizes, int n_in,
                              void* d_out, int out_size, void* d_ws, size_t ws_size,
                              hipStream_t stream) {
  const float* x      = (const float*)d_in[0];
  const int*   ei     = (const int*)d_in[1];
  const int*   batch  = (const int*)d_in[2];
  const float* W1     = (const float*)d_in[3];
  const float* a_src1 = (const float*)d_in[4];
  const float* a_dst1 = (const float*)d_in[5];
  const float* b1     = (const float*)d_in[6];
  const float* bn1_g  = (const float*)d_in[7];
  const float* bn1_b  = (const float*)d_in[8];
  const float* bn1_m  = (const float*)d_in[9];
  const float* bn1_v  = (const float*)d_in[10];
  const float* ln1_g  = (const float*)d_in[11];
  const float* ln1_b  = (const float*)d_in[12];
  const float* W2     = (const float*)d_in[13];
  const float* a_src2 = (const float*)d_in[14];
  const float* a_dst2 = (const float*)d_in[15];
  const float* b2     = (const float*)d_in[16];
  const float* bn2_g  = (const float*)d_in[17];
  const float* bn2_b  = (const float*)d_in[18];
  const float* bn2_m  = (const float*)d_in[19];
  const float* bn2_v  = (const float*)d_in[20];
  const float* ln2_g  = (const float*)d_in[21];
  const float* ln2_b  = (const float*)d_in[22];

  const int N  = in_sizes[0] / FW;       // 50000
  const int E  = in_sizes[1] / 2;        // 800000
  const int Bg = out_size / 512;         // 128
  const long ENt = (long)E + N;

  // ---- workspace carve ----
  char* ws = (char*)d_ws; size_t off = 0;
  auto carve = [&](size_t bytes) -> void* {
    void* p = ws + off; off += (bytes + 255) & ~(size_t)255; return p;
  };
  unsigned short* xbf   = (unsigned short*)carve((size_t)N * FW * 2);
  unsigned short* w1t   = (unsigned short*)carve((size_t)FW * FW * 2);       // [256 x 256]
  unsigned short* w2t   = (unsigned short*)carve((size_t)FW * 512 * 2);      // [256 x 512]
  float* hbuf           = (float*)carve((size_t)N * FW * 4);
  float* als            = (float*)carve((size_t)N * NHEAD * 4);
  float* ald            = (float*)carve((size_t)N * NHEAD * 4);
  float* ebuf           = (float*)carve((size_t)ENt * NHEAD * 4);
  float* emax           = (float*)carve((size_t)N * NHEAD * 4);
  float* denom          = (float*)carve((size_t)N * NHEAD * 4);
  float* agg            = (float*)carve((size_t)N * FW * 4);
  float* post1          = (float*)carve((size_t)N * FW * 4);
  unsigned short* h2in  = (unsigned short*)carve((size_t)N * 512 * 2);
  int* root_idx         = (int*)carve((size_t)Bg * 4);
  int* ron              = (int*)carve((size_t)N * 4);
  float* counts         = (float*)carve((size_t)Bg * 4);
  (void)ws_size;

  float* post2 = agg;   // post kernel may run in place

  auto cdiv = [](long a, long b) -> int { return (int)((a + b - 1) / b); };
  const int TB = 256;

  // ---- prep: bf16 conversions / weight transposes ----
  conv_bf16_kernel<<<cdiv((long)N * FW, TB), TB, 0, stream>>>(x, xbf, (long)N * FW);
  transpose_bf16_kernel<<<cdiv((long)FW * FW, TB), TB, 0, stream>>>(W1, w1t, FW, FW);
  transpose_bf16_kernel<<<cdiv((long)512 * FW, TB), TB, 0, stream>>>(W2, w2t, 512, FW);

  const int tiles = (N / 16) * (FW / 64);          // 16x64 strip per wave
  const int gemm_blocks = cdiv(tiles, TB / 32);    // 8 waves/block

  // ================= Layer 1 =================
  gemm_bf16_wmma_kernel<<<gemm_blocks, TB, 0, stream>>>(
      (const bf16_t*)xbf, (const bf16_t*)w1t, hbuf, N, FW, FW);
  logits_kernel<<<cdiv((long)N * NHEAD, TB), TB, 0, stream>>>(
      hbuf, a_src1, a_dst1, als, ald, N);
  fill_f32_kernel<<<cdiv((long)N * NHEAD, TB), TB, 0, stream>>>(
      emax, -__builtin_huge_valf(), (long)N * NHEAD);
  fill_f32_kernel<<<cdiv((long)N * NHEAD, TB), TB, 0, stream>>>(
      denom, 0.f, (long)N * NHEAD);
  fill_f32_kernel<<<cdiv((long)N * FW, TB), TB, 0, stream>>>(
      agg, 0.f, (long)N * FW);
  edge_logit_max_kernel<<<cdiv(ENt * NHEAD, TB), TB, 0, stream>>>(
      ei, als, ald, ebuf, emax, E, N);
  edge_exp_sum_kernel<<<cdiv(ENt * NHEAD, TB), TB, 0, stream>>>(
      ei, ebuf, emax, denom, E, N);
  edge_aggregate_kernel<<<cdiv(ENt, TB / 32), TB, 0, stream>>>(
      ei, ebuf, denom, hbuf, agg, E, N);
  post_kernel<<<N, FW, 0, stream>>>(
      agg, b1, bn1_g, bn1_b, bn1_m, bn1_v, ln1_g, ln1_b, post1);

  // ---- root extend ----
  root_detect_kernel<<<cdiv(N, TB), TB, 0, stream>>>(batch, root_idx, N);
  root_gather_kernel<<<cdiv(N, TB), TB, 0, stream>>>(batch, root_idx, ron, N);
  build_h2in_kernel<<<cdiv((long)N * 512, TB), TB, 0, stream>>>(
      post1, xbf, ron, h2in, N);

  // ================= Layer 2 =================
  gemm_bf16_wmma_kernel<<<gemm_blocks, TB, 0, stream>>>(
      (const bf16_t*)h2in, (const bf16_t*)w2t, hbuf, N, 512, FW);
  logits_kernel<<<cdiv((long)N * NHEAD, TB), TB, 0, stream>>>(
      hbuf, a_src2, a_dst2, als, ald, N);
  fill_f32_kernel<<<cdiv((long)N * NHEAD, TB), TB, 0, stream>>>(
      emax, -__builtin_huge_valf(), (long)N * NHEAD);
  fill_f32_kernel<<<cdiv((long)N * NHEAD, TB), TB, 0, stream>>>(
      denom, 0.f, (long)N * NHEAD);
  fill_f32_kernel<<<cdiv((long)N * FW, TB), TB, 0, stream>>>(
      agg, 0.f, (long)N * FW);
  edge_logit_max_kernel<<<cdiv(ENt * NHEAD, TB), TB, 0, stream>>>(
      ei, als, ald, ebuf, emax, E, N);
  edge_exp_sum_kernel<<<cdiv(ENt * NHEAD, TB), TB, 0, stream>>>(
      ei, ebuf, emax, denom, E, N);
  edge_aggregate_kernel<<<cdiv(ENt, TB / 32), TB, 0, stream>>>(
      ei, ebuf, denom, hbuf, agg, E, N);
  post_kernel<<<N, FW, 0, stream>>>(
      agg, b2, bn2_g, bn2_b, bn2_m, bn2_v, ln2_g, ln2_b, post2);

  // ---- final segment mean: [B, 512] ----
  float* outp = (float*)d_out;
  fill_f32_kernel<<<cdiv((long)Bg * 512, TB), TB, 0, stream>>>(
      outp, 0.f, (long)Bg * 512);
  fill_f32_kernel<<<cdiv((long)Bg, TB), TB, 0, stream>>>(counts, 0.f, (long)Bg);
  final_accum_kernel<<<cdiv((long)N * 512, TB), TB, 0, stream>>>(
      post2, post1, ron, batch, outp, counts, N);
  final_div_kernel<<<cdiv((long)Bg * 512, TB), TB, 0, stream>>>(outp, counts, Bg);
}
